// Attention_86062554677532
// MI455X (gfx1250) — matrix operand order repaired and checked by reading, not compiled, not run on hardware
//
#include <hip/hip_runtime.h>
#include <stdint.h>

// Problem constants (from reference): B=2,H=16,S=2048,D=64
#define BDIM 2
#define HDIM 16
#define SDIM 2048
#define DDIM 64
#define KC   128      // keys staged in LDS per chunk
#define QTILE 128     // query rows per workgroup
#define NWAVES 8      // 256 threads / wave32

typedef __attribute__((ext_vector_type(16))) __bf16 v16bf;
typedef __attribute__((ext_vector_type(8)))  float  v8f;

union BF16x16 { v16bf v; uint4 q[2]; unsigned w[8]; unsigned short u[16]; };

// round-half-up f32->bf16 (cheap: one v_add); scalar form
__device__ __forceinline__ unsigned short cvt_bf16(float f) {
  return (unsigned short)((__float_as_uint(f) + 0x8000u) >> 16);
}
// pack two f32 -> two bf16 in one dword: 2 v_add + 1 v_perm_b32
__device__ __forceinline__ unsigned pack_bf16x2(float lo, float hi) {
  unsigned a = __float_as_uint(lo) + 0x8000u;
  unsigned b = __float_as_uint(hi) + 0x8000u;
  return __builtin_amdgcn_perm(b, a, 0x07060302u);  // {b.hi16, a.hi16}
}

__device__ __forceinline__ v8f wmma_bf16(v16bf a, v16bf b, v8f c) {
  // (neg_a, A, neg_b, B, c_mod, C, reuse_a, reuse_b)
  return __builtin_amdgcn_wmma_f32_16x16x32_bf16(false, a, false, b, (short)0, c,
                                                 false, false);
}

// ---- fragment loaders -------------------------------------------------------
// A (16x32 bf16, MxK): lane<16 -> row=lane, K = {k0..k0+7, k0+16..k0+23}
//                      lane>=16 -> same row set, K shifted by +8
// `scale` is a power of two (exact): folds the softmax 1/sqrt(d) factors here.
__device__ __forceinline__ v16bf load_a_global(const float* mat, int row, int k0,
                                               float scale) {
  const int lane = threadIdx.x & 31;
  const float* rp = mat + (size_t)row * DDIM + k0 + (((lane >> 4) & 1) ? 8 : 0);
  BF16x16 r;
#pragma unroll
  for (int j = 0; j < 4; ++j)
    r.w[j] = pack_bf16x2(rp[2 * j] * scale, rp[2 * j + 1] * scale);
#pragma unroll
  for (int j = 0; j < 4; ++j)
    r.w[4 + j] = pack_bf16x2(rp[16 + 2 * j] * scale, rp[16 + 2 * j + 1] * scale);
  return r.v;
}

// B (32x16 bf16, KxN) from row-major LDS [key][DDIM]:
// lane: N = lane&15 (key = keyBase+N), K = dBase + {kh..kh+7, kh+16..kh+23}, kh = lane<16 ? 0 : 8
__device__ __forceinline__ v16bf ldsB_score(const unsigned short* lds, int keyBase, int dBase) {
  const int lane = threadIdx.x & 31;
  const unsigned short* p =
      lds + (size_t)(keyBase + (lane & 15)) * DDIM + dBase + (((lane >> 4) & 1) ? 8 : 0);
  BF16x16 r;
  r.q[0] = *(const uint4*)(p);
  r.q[1] = *(const uint4*)(p + 16);
  return r.v;
}

// B (32x16 bf16) for P*V from transposed LDS V [d][KC]:
// lane: N = d = dBase + (lane&15); K = keys keyBase + {kh..kh+7, kh+16..kh+23}, kh = lane<16 ? 0 : 8
__device__ __forceinline__ v16bf ldsB_v(const unsigned short* vls, int keyBase, int dBase) {
  const int lane = threadIdx.x & 31;
  const unsigned short* p =
      vls + (size_t)(dBase + (lane & 15)) * KC + keyBase + (((lane >> 4) & 1) ? 8 : 0);
  BF16x16 r;
  r.q[0] = *(const uint4*)(p);
  r.q[1] = *(const uint4*)(p + 16);
  return r.v;
}

// A (16x32 bf16) of P from per-wave row-major scratch [16][32]
__device__ __forceinline__ v16bf load_aP(const unsigned short* ps) {
  const int lane = threadIdx.x & 31;
  const unsigned short* p = ps + (size_t)(lane & 15) * 32 + (((lane >> 4) & 1) ? 8 : 0);
  BF16x16 r;
  r.q[0] = *(const uint4*)(p);        // K = kh .. kh+7
  r.q[1] = *(const uint4*)(p + 16);   // K = kh+16 .. kh+23
  return r.v;
}

// ---- cooperative chunk staging ---------------------------------------------
__device__ __forceinline__ void load_chunk(unsigned short* dst, const float* src) {
  const int t = threadIdx.x;
#pragma unroll
  for (int j = 0; j < 8; ++j) {
    int f = t + j * 256;            // float4 index in [0, 2048)
    int key = f >> 4;               // 16 float4 per 64-wide row
    int dq = (f & 15) << 2;
    float4 x = *(const float4*)(src + (size_t)key * DDIM + dq);
    uint2 w;
    w.x = pack_bf16x2(x.x, x.y);
    w.y = pack_bf16x2(x.z, x.w);
    *(uint2*)(dst + (size_t)key * DDIM + dq) = w;
  }
}

__device__ __forceinline__ void load_chunk_vT(unsigned short* dst /*[DDIM][KC]*/,
                                              const float* src) {
  const int t = threadIdx.x;
#pragma unroll
  for (int j = 0; j < 8; ++j) {
    int f = t + j * 256;
    int key = f >> 4;
    int dq = (f & 15) << 2;
    float4 x = *(const float4*)(src + (size_t)key * DDIM + dq);
    dst[(size_t)(dq + 0) * KC + key] = cvt_bf16(x.x);
    dst[(size_t)(dq + 1) * KC + key] = cvt_bf16(x.y);
    dst[(size_t)(dq + 2) * KC + key] = cvt_bf16(x.z);
    dst[(size_t)(dq + 3) * KC + key] = cvt_bf16(x.w);
  }
}

// ---- cross-lane reductions within 16-lane halves (used ONCE after pass 1) ---
__device__ __forceinline__ float redmax16(float v) {
#pragma unroll
  for (int m = 1; m <= 8; m <<= 1) v = fmaxf(v, __shfl_xor(v, m, 32));
  return v;
}
__device__ __forceinline__ float redsum16(float v) {
#pragma unroll
  for (int m = 1; m <= 8; m <<= 1) v += __shfl_xor(v, m, 32);
  return v;
}

// score tile (16 queries x 16 keys): s = (Q/8 . K^T) * (MQ/8 . MK^T)
// (the 1/8 factors are pre-folded into the A fragments)
__device__ __forceinline__ v8f score_tile(const unsigned short* kls,
                                          const unsigned short* mkls, int sub,
                                          v16bf aQ0, v16bf aQ1, v16bf aM0, v16bf aM1) {
  v16bf bk0 = ldsB_score(kls, sub * 16, 0);
  v16bf bk1 = ldsB_score(kls, sub * 16, 32);
  v16bf bm0 = ldsB_score(mkls, sub * 16, 0);
  v16bf bm1 = ldsB_score(mkls, sub * 16, 32);
  v8f c = {};
  c = wmma_bf16(aQ0, bk0, c);
  c = wmma_bf16(aQ1, bk1, c);
  v8f cm = {};
  cm = wmma_bf16(aM0, bm0, cm);
  cm = wmma_bf16(aM1, bm1, cm);
  v8f s;
#pragma unroll
  for (int i = 0; i < 8; ++i) s[i] = c[i] * cm[i];
  return s;
}

__global__ __launch_bounds__(256)
__attribute__((amdgpu_waves_per_eu(4)))  // cap VGPRs at 256: no MSB switching, >=4 waves/SIMD
void masked_attention_kernel(const float* __restrict__ Q, const float* __restrict__ K,
                             const float* __restrict__ V, const float* __restrict__ MQ,
                             const float* __restrict__ MK, float* __restrict__ pval,
                             float* __restrict__ pattn) {
  __shared__ unsigned short kls[KC * DDIM];        // 16 KB  K chunk  [key][d]
  __shared__ unsigned short mkls[KC * DDIM];       // 16 KB  MK chunk [key][d]
  __shared__ unsigned short vls[DDIM * KC];        // 16 KB  V chunk  [d][key] (transposed)
  __shared__ unsigned short pscr[NWAVES][16 * 32]; //  8 KB  per-wave P transpose scratch

  const int lane = threadIdx.x & 31;
  const int wave = threadIdx.x >> 5;
  const int bh = blockIdx.x >> 4;            // 0..31
  const int q0 = (blockIdx.x & 15) * QTILE;  // query tile base
  const int wq0 = q0 + wave * 16;            // this wave's 16 query rows
  const int arow = wq0 + (lane & 15);

  const size_t moff = (size_t)bh * SDIM * DDIM;
  const float* Qb = Q + moff;
  const float* Kb = K + moff;
  const float* Vb = V + moff;
  const float* MQb = MQ + moff;
  const float* MKb = MK + moff;

  // Q / MQ A-fragments (pre-scaled by 1/sqrt(D) = 1/8, exact power of two)
  const v16bf aQ0 = load_a_global(Qb, arow, 0, 0.125f);
  const v16bf aQ1 = load_a_global(Qb, arow, 32, 0.125f);
  const v16bf aM0 = load_a_global(MQb, arow, 0, 0.125f);
  const v16bf aM1 = load_a_global(MQb, arow, 32, 0.125f);

  // Streaming softmax with PER-LANE accumulators: each lane tracks max / sumexp
  // over the columns it owns; a single cross-lane merge happens after pass 1.
  float mL[8], lL[8];
#pragma unroll
  for (int i = 0; i < 8; ++i) { mL[i] = -3.0e38f; lL[i] = 0.0f; }

  // ---------------- pass 1: per-lane online max / sum-exp ----------------
  for (int kc = 0; kc < SDIM; kc += KC) {
    if (kc + KC < SDIM) {  // one 128B line per thread of the next chunk
      __builtin_prefetch(Kb + (size_t)(kc + KC) * DDIM + threadIdx.x * 32, 0, 1);
      __builtin_prefetch(MKb + (size_t)(kc + KC) * DDIM + threadIdx.x * 32, 0, 1);
    }
    load_chunk(kls, Kb + (size_t)kc * DDIM);
    load_chunk(mkls, MKb + (size_t)kc * DDIM);
    __syncthreads();
#pragma unroll 4    // 4 tiles in flight: enough ILP, keeps allocation under 256 VGPRs
    for (int sub = 0; sub < KC / 16; ++sub) {
      v8f s = score_tile(kls, mkls, sub, aQ0, aQ1, aM0, aM1);
#pragma unroll
      for (int i = 0; i < 8; ++i) {
        // One-exp online update: of {exp(m-nm), exp(s-nm)} exactly one is
        // exp(-|s-m|), the other is 1.  -|d| folds into v_exp src modifiers.
        float d = s[i] - mL[i];
        float e = __expf(-fabsf(d));
        bool gt = d > 0.0f;
        lL[i] = fmaf(lL[i], gt ? e : 1.0f, gt ? 1.0f : e);
        mL[i] = fmaxf(mL[i], s[i]);
      }
    }
    __syncthreads();
  }

  // single cross-lane merge: 8 shuffles per row, once per kernel
  float m[8], linv[8];
#pragma unroll
  for (int i = 0; i < 8; ++i) {
    float nm = redmax16(mL[i]);
    float l = redsum16(lL[i] * __expf(mL[i] - nm));
    m[i] = nm;
    linv[i] = 1.0f / l;
  }

  v8f accO[4] = {};

  // ---------------- pass 2: write p_attn once, accumulate P*V ----------------
  for (int kc = 0; kc < SDIM; kc += KC) {
    if (kc + KC < SDIM) {
      __builtin_prefetch(Kb + (size_t)(kc + KC) * DDIM + threadIdx.x * 32, 0, 1);
      __builtin_prefetch(MKb + (size_t)(kc + KC) * DDIM + threadIdx.x * 32, 0, 1);
      __builtin_prefetch(Vb + (size_t)(kc + KC) * DDIM + threadIdx.x * 32, 0, 1);
    }
    load_chunk(kls, Kb + (size_t)kc * DDIM);
    load_chunk(mkls, MKb + (size_t)kc * DDIM);
    load_chunk_vT(vls, Vb + (size_t)kc * DDIM);
    __syncthreads();
#pragma unroll 2
    for (int sub32 = 0; sub32 < KC / 32; ++sub32) {
#pragma unroll
      for (int h = 0; h < 2; ++h) {
        const int sub = sub32 * 2 + h;
        v8f s = score_tile(kls, mkls, sub, aQ0, aQ1, aM0, aM1);
        const int col = kc + sub * 16 + (lane & 15);
#pragma unroll
        for (int i = 0; i < 8; ++i) {
          const int mloc = i + (((lane >> 4) & 1) << 3);  // C/D row for this lane
          float p = __expf(s[i] - m[i]) * linv[i];
          pattn[((size_t)bh * SDIM + (size_t)(wq0 + mloc)) * SDIM + col] = p;
          pscr[wave][mloc * 32 + h * 16 + (lane & 15)] = cvt_bf16(p);
        }
      }
      __builtin_amdgcn_wave_barrier();  // LDS transpose: cross-lane, same wave (DS in-order)
      v16bf aP = load_aP(&pscr[wave][0]);
      v16bf bV0 = ldsB_v(vls, sub32 * 32, 0);
      v16bf bV1 = ldsB_v(vls, sub32 * 32, 16);
      v16bf bV2 = ldsB_v(vls, sub32 * 32, 32);
      v16bf bV3 = ldsB_v(vls, sub32 * 32, 48);
      accO[0] = wmma_bf16(aP, bV0, accO[0]);
      accO[1] = wmma_bf16(aP, bV1, accO[1]);
      accO[2] = wmma_bf16(aP, bV2, accO[2]);
      accO[3] = wmma_bf16(aP, bV3, accO[3]);
    }
    __syncthreads();
  }

  // ---------------- write p_val ----------------
#pragma unroll
  for (int n = 0; n < 4; ++n) {
#pragma unroll
    for (int i = 0; i < 8; ++i) {
      const int mloc = i + (((lane >> 4) & 1) << 3);
      pval[((size_t)bh * SDIM + (size_t)(wq0 + mloc)) * DDIM + n * 16 + (lane & 15)] =
          accO[n][i];
    }
  }
}

extern "C" void kernel_launch(void* const* d_in, const int* in_sizes, int n_in,
                              void* d_out, int out_size, void* d_ws, size_t ws_size,
                              hipStream_t stream) {
  (void)in_sizes; (void)n_in; (void)out_size; (void)d_ws; (void)ws_size;
  const float* Q  = (const float*)d_in[0];
  const float* K  = (const float*)d_in[1];
  const float* V  = (const float*)d_in[2];
  const float* MQ = (const float*)d_in[3];
  const float* MK = (const float*)d_in[4];
  float* pval  = (float*)d_out;
  float* pattn = pval + (size_t)BDIM * HDIM * SDIM * DDIM;  // outputs concatenated
  dim3 grid(BDIM * HDIM * (SDIM / QTILE));  // 512 workgroups
  masked_attention_kernel<<<grid, 256, 0, stream>>>(Q, K, V, MQ, MK, pval, pattn);
}
